// VGGProgressiveReplacementGraspNet_16733192585531
// MI455X (gfx1250) — matrix operand order repaired and checked by reading, not compiled, not run on hardware
//
#include <hip/hip_runtime.h>
#include <hip/hip_bf16.h>

// ---------------------------------------------------------------------------
// Shapes (fixed by the reference): B=4, N=20000, Ns=1024, K=8192, Dv=768, Dc=256
// ---------------------------------------------------------------------------
#define BATCH 4
#define NS    1024
#define KPTS  8192
#define DV    768
#define DC    256
#define NT    32            // output columns per workgroup in the fused kernel

typedef __attribute__((ext_vector_type(16))) __bf16 v16bf;
typedef __attribute__((ext_vector_type(8)))  __bf16 v8bf;
typedef __attribute__((ext_vector_type(8)))  float  v8f;

union FragU { v16bf v; v8bf h[2]; };

// ---------------------------------------------------------------------------
// WMMA helpers (CDNA5 16x16x32 bf16 -> f32), contiguous k order (k = 16h + s) in both A and B.
// ---------------------------------------------------------------------------
__device__ __forceinline__ v8f wmma_bf16(v16bf a, v16bf b, v8f c) {
    // 8 args: (neg_a, A, neg_b, B, c_mod, C, reuse_a, reuse_b)
    return __builtin_amdgcn_wmma_f32_16x16x32_bf16(false, a, false, b,
                                                   (short)0, c, false, false);
}

// A fragment: 16(M) x 32(K), source row-major with row stride ldk.
// lanes 0-15: M=lane, K = kk..kk+15
// lanes 16-31: M=lane-16, K = kk+16..kk+31
__device__ __forceinline__ v16bf load_frag_A(const __bf16* __restrict__ base,
                                             int row0, int kk, int ldk, int lane) {
    const int hi = lane >> 4;
    const int m  = lane & 15;
    const __bf16* p = base + (size_t)(row0 + m) * ldk + kk + hi * 16;
    FragU f;
    f.h[0] = *(const v8bf*)(p);
    f.h[1] = *(const v8bf*)(p + 8);
    return f.v;
}

// B fragment: 32(K) x 16(N), source column-major: column n contiguous in K,
// column stride ldcol elements.
// lanes 0-15: N=lane, K=kk..kk+15 ; lanes 16-31: N=lane-16, K=kk+16..kk+31
__device__ __forceinline__ v16bf load_frag_B(const __bf16* __restrict__ base,
                                             int col0, int kk, int ldcol, int lane) {
    const int hi = lane >> 4;
    const int n  = lane & 15;
    const __bf16* p = base + (size_t)(col0 + n) * ldcol + kk + hi * 16;
    FragU f;
    f.h[0] = *(const v8bf*)(p);
    f.h[1] = *(const v8bf*)(p + 8);
    return f.v;
}

// ---------------------------------------------------------------------------
// K1: nearest-neighbor argmin.  Normalization (centroid/scale) is a per-batch
// strictly-monotone transform of d^2, so argmin over RAW coordinates is
// identical -> point_cloud is never read.
// grid (NS/256, BATCH), block 256, one seed per thread.
// ---------------------------------------------------------------------------
__global__ __launch_bounds__(256) void nn_kernel(const float* __restrict__ seed,
                                                 const float* __restrict__ pts,
                                                 int* __restrict__ idx) {
    __shared__ float tile[1024 * 3];
    const int b = blockIdx.y;
    const int n = blockIdx.x * 256 + threadIdx.x;

    const float sx = seed[((size_t)b * NS + n) * 3 + 0];
    const float sy = seed[((size_t)b * NS + n) * 3 + 1];
    const float sz = seed[((size_t)b * NS + n) * 3 + 2];

    float best = 3.4e38f;
    int   bi   = 0;
    for (int k0 = 0; k0 < KPTS; k0 += 1024) {
        __syncthreads();
        for (int j = threadIdx.x; j < 1024 * 3; j += 256)
            tile[j] = pts[((size_t)b * KPTS + k0) * 3 + j];
        __syncthreads();
        #pragma unroll 4
        for (int k = 0; k < 1024; ++k) {
            const float dx = sx - tile[k * 3 + 0];
            const float dy = sy - tile[k * 3 + 1];
            const float dz = sz - tile[k * 3 + 2];
            const float d  = dx * dx + dy * dy + dz * dz;
            if (d < best) { best = d; bi = k0 + k; }
        }
    }
    idx[b * NS + n] = bi;
}

// ---------------------------------------------------------------------------
// K2: f32 -> bf16 weight cast (one-time; weights then L2-resident).
// ---------------------------------------------------------------------------
__global__ void cast_bf16(const float* __restrict__ src, __bf16* __restrict__ dst, int n) {
    const int i = blockIdx.x * blockDim.x + threadIdx.x;
    if (i < n) dst[i] = (__bf16)src[i];
}

// ---------------------------------------------------------------------------
// K3: gather feature columns at idx, cast to bf16, store column-contiguous:
// Xg[b][n][c] so WMMA B fragments become contiguous b128 loads.
// grid (NS, BATCH), block 256.
// ---------------------------------------------------------------------------
__global__ __launch_bounds__(256) void gather_cast(const float* __restrict__ feat,
                                                   const int* __restrict__ idx,
                                                   __bf16* __restrict__ Xg) {
    const int b  = blockIdx.y;
    const int n  = blockIdx.x;
    const int id = idx[b * NS + n];
    const float* src = feat + (size_t)b * DV * KPTS + id;
    __bf16* dst = Xg + ((size_t)b * NS + n) * DV;
    for (int c = threadIdx.x; c < DV; c += blockDim.x)
        dst[c] = (__bf16)src[(size_t)c * KPTS];
}

// ---------------------------------------------------------------------------
// K4: fused proj-GEMM + LayerNorm + adapter (GEMM-ReLU-GEMM) + blend.
// grid (NS/NT, BATCH), block 256 (= 8 wave32).
// Wave w owns rows [32w, 32w+32) as 2x2 tiles of 16x16 WMMA accumulators.
// ---------------------------------------------------------------------------
__global__ __launch_bounds__(256) void adapter_kernel(
    const __bf16* __restrict__ Xg,   // [B][NS][DV] gathered bf16 columns
    const __bf16* __restrict__ Wp,   // [DC][DV]
    const __bf16* __restrict__ Wa1,  // [DC][DC]
    const __bf16* __restrict__ Wa2,  // [DC][DC]
    const float*  __restrict__ bp,
    const float*  __restrict__ ba1,
    const float*  __restrict__ ba2,
    const float*  __restrict__ fg,   // [B][DC][NS]
    float*        __restrict__ out)  // [B][DC][NS]
{
    __shared__ __bf16 xb[NT][DC];    // x (pre/post LN), column-major, 16 KB
    __shared__ __bf16 hb[NT][DC];    // relu(h), column-major, 16 KB
    __shared__ float  red_s[8][NT];
    __shared__ float  red_q[8][NT];
    __shared__ float  mu_s[NT];
    __shared__ float  ri_s[NT];

    const int b    = blockIdx.y;
    const int n0   = blockIdx.x * NT;
    const int t    = threadIdx.x;
    const int lane = t & 31;
    const int wave = t >> 5;
    const int hi   = lane >> 4;
    const int l16  = lane & 15;

    const __bf16* Xcol = Xg + ((size_t)b * NS + n0) * DV;

    // ---------------- GEMM1: x = W_proj @ gathered + b_proj  (K = 768) ------
    v8f acc[2][2];
    acc[0][0] = {}; acc[0][1] = {}; acc[1][0] = {}; acc[1][1] = {};
    for (int kk = 0; kk < DV; kk += 32) {
        const v16bf a0 = load_frag_A(Wp, wave * 32,      kk, DV, lane);
        const v16bf a1 = load_frag_A(Wp, wave * 32 + 16, kk, DV, lane);
        const v16bf b0 = load_frag_B(Xcol, 0,  kk, DV, lane);
        const v16bf b1 = load_frag_B(Xcol, 16, kk, DV, lane);
        acc[0][0] = wmma_bf16(a0, b0, acc[0][0]);
        acc[0][1] = wmma_bf16(a0, b1, acc[0][1]);
        acc[1][0] = wmma_bf16(a1, b0, acc[1][0]);
        acc[1][1] = wmma_bf16(a1, b1, acc[1][1]);
    }
    // bias + spill pre-LN x to LDS (bf16, column-major)
    #pragma unroll
    for (int mi = 0; mi < 2; ++mi)
        #pragma unroll
        for (int ni = 0; ni < 2; ++ni)
            #pragma unroll
            for (int r = 0; r < 8; ++r) {
                const int row = wave * 32 + mi * 16 + hi * 8 + r;
                const int col = ni * 16 + l16;
                xb[col][row] = (__bf16)(acc[mi][ni][r] + bp[row]);
            }
    __syncthreads();

    // ---------------- LayerNorm over the 256 channels of each column -------
    {
        const int col = t & (NT - 1);
        const int seg = t >> 5;          // 8 segments of 32 rows
        float s = 0.f, q = 0.f;
        #pragma unroll 8
        for (int r = 0; r < 32; ++r) {
            const float v = (float)xb[col][seg * 32 + r];
            s += v; q += v * v;
        }
        red_s[seg][col] = s; red_q[seg][col] = q;
    }
    __syncthreads();
    if (t < NT) {
        float s = 0.f, q = 0.f;
        #pragma unroll
        for (int g = 0; g < 8; ++g) { s += red_s[g][t]; q += red_q[g][t]; }
        const float mu  = s * (1.0f / DC);
        const float var = q * (1.0f / DC) - mu * mu;
        mu_s[t] = mu;
        ri_s[t] = rsqrtf(var + 1e-5f);
    }
    __syncthreads();
    {
        const int col = t & (NT - 1);
        const int seg = t >> 5;
        const float mu = mu_s[col], ri = ri_s[col];
        #pragma unroll 8
        for (int r = 0; r < 32; ++r) {
            const int row = seg * 32 + r;
            xb[col][row] = (__bf16)(((float)xb[col][row] - mu) * ri);
        }
    }
    __syncthreads();

    // ---------------- GEMM2: h = relu(W_a1 @ x_ln + b_a1)  (K = 256) -------
    acc[0][0] = {}; acc[0][1] = {}; acc[1][0] = {}; acc[1][1] = {};
    for (int kk = 0; kk < DC; kk += 32) {
        const v16bf a0 = load_frag_A(Wa1, wave * 32,      kk, DC, lane);
        const v16bf a1 = load_frag_A(Wa1, wave * 32 + 16, kk, DC, lane);
        const v16bf b0 = load_frag_B(&xb[0][0], 0,  kk, DC, lane);
        const v16bf b1 = load_frag_B(&xb[0][0], 16, kk, DC, lane);
        acc[0][0] = wmma_bf16(a0, b0, acc[0][0]);
        acc[0][1] = wmma_bf16(a0, b1, acc[0][1]);
        acc[1][0] = wmma_bf16(a1, b0, acc[1][0]);
        acc[1][1] = wmma_bf16(a1, b1, acc[1][1]);
    }
    #pragma unroll
    for (int mi = 0; mi < 2; ++mi)
        #pragma unroll
        for (int ni = 0; ni < 2; ++ni)
            #pragma unroll
            for (int r = 0; r < 8; ++r) {
                const int row = wave * 32 + mi * 16 + hi * 8 + r;
                const int col = ni * 16 + l16;
                float v = acc[mi][ni][r] + ba1[row];
                hb[col][row] = (__bf16)(v > 0.f ? v : 0.f);
            }
    __syncthreads();

    // ---------------- GEMM3 + blend:  out = 0.5*fg + 0.5*(W_a2 @ h + b_a2) -
    acc[0][0] = {}; acc[0][1] = {}; acc[1][0] = {}; acc[1][1] = {};
    for (int kk = 0; kk < DC; kk += 32) {
        const v16bf a0 = load_frag_A(Wa2, wave * 32,      kk, DC, lane);
        const v16bf a1 = load_frag_A(Wa2, wave * 32 + 16, kk, DC, lane);
        const v16bf b0 = load_frag_B(&hb[0][0], 0,  kk, DC, lane);
        const v16bf b1 = load_frag_B(&hb[0][0], 16, kk, DC, lane);
        acc[0][0] = wmma_bf16(a0, b0, acc[0][0]);
        acc[0][1] = wmma_bf16(a0, b1, acc[0][1]);
        acc[1][0] = wmma_bf16(a1, b0, acc[1][0]);
        acc[1][1] = wmma_bf16(a1, b1, acc[1][1]);
    }
    #pragma unroll
    for (int mi = 0; mi < 2; ++mi)
        #pragma unroll
        for (int ni = 0; ni < 2; ++ni)
            #pragma unroll
            for (int r = 0; r < 8; ++r) {
                const int row  = wave * 32 + mi * 16 + hi * 8 + r;
                const int col  = n0 + ni * 16 + l16;
                const size_t o = ((size_t)b * DC + row) * NS + col;
                out[o] = 0.5f * fg[o] + 0.5f * (acc[mi][ni][r] + ba2[row]);
            }
}

// ---------------------------------------------------------------------------
// Host launcher.
// Input order: 0 point_cloud (unused: argmin invariant under normalization),
// 1 seed_xyz, 2 pts, 3 feat, 4 f_graspnet, 5 W_proj, 6 b_proj, 7 W_a1,
// 8 b_a1, 9 W_a2, 10 b_a2.
// ---------------------------------------------------------------------------
extern "C" void kernel_launch(void* const* d_in, const int* in_sizes, int n_in,
                              void* d_out, int out_size, void* d_ws, size_t ws_size,
                              hipStream_t stream) {
    (void)in_sizes; (void)n_in; (void)out_size; (void)ws_size;

    const float* seed  = (const float*)d_in[1];
    const float* pts   = (const float*)d_in[2];
    const float* feat  = (const float*)d_in[3];
    const float* fg    = (const float*)d_in[4];
    const float* Wp_f  = (const float*)d_in[5];
    const float* bp    = (const float*)d_in[6];
    const float* Wa1_f = (const float*)d_in[7];
    const float* ba1   = (const float*)d_in[8];
    const float* Wa2_f = (const float*)d_in[9];
    const float* ba2   = (const float*)d_in[10];
    float* out = (float*)d_out;

    // Workspace layout (bytes)
    char* ws = (char*)d_ws;
    int*    idx = (int*)ws;                                   //  16 KB
    __bf16* Wp  = (__bf16*)(ws + 16384);                      // 384 KB
    __bf16* Wa1 = (__bf16*)(ws + 16384 + 393216);             // 128 KB
    __bf16* Wa2 = (__bf16*)(ws + 16384 + 393216 + 131072);    // 128 KB
    __bf16* Xg  = (__bf16*)(ws + 16384 + 393216 + 2*131072);  // 6 MB

    // 1) nearest-neighbor indices
    nn_kernel<<<dim3(NS / 256, BATCH), 256, 0, stream>>>(seed, pts, idx);

    // 2) weight casts (f32 -> bf16)
    cast_bf16<<<(DC * DV + 255) / 256, 256, 0, stream>>>(Wp_f,  Wp,  DC * DV);
    cast_bf16<<<(DC * DC + 255) / 256, 256, 0, stream>>>(Wa1_f, Wa1, DC * DC);
    cast_bf16<<<(DC * DC + 255) / 256, 256, 0, stream>>>(Wa2_f, Wa2, DC * DC);

    // 3) gather feature columns, column-contiguous bf16
    gather_cast<<<dim3(NS, BATCH), 256, 0, stream>>>(feat, idx, Xg);

    // 4) fused GEMM + LN + GEMM-ReLU-GEMM + blend
    adapter_kernel<<<dim3(NS / NT, BATCH), 256, 0, stream>>>(
        Xg, Wp, Wa1, Wa2, bp, ba1, ba2, fg, out);
}
